// Set_Abstraction_Module_53764400611866
// MI455X (gfx1250) — matrix operand order repaired and checked by reading, not compiled, not run on hardware
//
#include <hip/hip_runtime.h>
#include <math.h>

typedef _Float16 half_t;
typedef __attribute__((ext_vector_type(16))) _Float16 v16h;
typedef __attribute__((ext_vector_type(8)))  _Float16 v8h;
typedef __attribute__((ext_vector_type(8)))  float    v8f;

#define BB 8
#define NN 4096
#define CFEAT 64
#define MM 2048
#define KK 32
#define RR (BB*MM*KK)   /* 524288 rows of the (b,m,k) tensor */

// ---------------------------------------------------------------------------
// Farthest point sampling: one workgroup per batch, LDS-resident distances,
// two-level argmax with JAX argmax tie-break (lowest index wins).
// ---------------------------------------------------------------------------
__global__ __launch_bounds__(1024)
void fps_kernel(const float* __restrict__ pose, int* __restrict__ fps_idx) {
  const int b = blockIdx.x;
  const int tid = threadIdx.x;
  __shared__ float dist_s[NN];
  __shared__ float rv[1024];
  __shared__ int   ri[1024];
  const float* P = pose + (size_t)b * NN * 3;
  for (int j = tid; j < NN; j += 1024) dist_s[j] = 1e10f;
  if (tid == 0) fps_idx[b * MM] = 0;
  int last = 0;
  __syncthreads();
  for (int step = 1; step < MM; ++step) {
    const float lx = P[last*3+0], ly = P[last*3+1], lz = P[last*3+2];
    float best = -1.0f; int bi = 0;
    for (int j = tid; j < NN; j += 1024) {
      float dx = P[j*3+0]-lx, dy = P[j*3+1]-ly, dz = P[j*3+2]-lz;
      float d2 = dx*dx + dy*dy + dz*dz;
      float dm = fminf(dist_s[j], d2);
      dist_s[j] = dm;
      if (dm > best) { best = dm; bi = j; }   // strict > keeps lowest j
    }
    rv[tid] = best; ri[tid] = bi;
    __syncthreads();
    for (int off = 512; off > 0; off >>= 1) {
      if (tid < off) {
        float v2 = rv[tid+off]; int i2 = ri[tid+off];
        if (v2 > rv[tid] || (v2 == rv[tid] && i2 < ri[tid])) { rv[tid] = v2; ri[tid] = i2; }
      }
      __syncthreads();
    }
    last = ri[0];
    if (tid == 0) fps_idx[b*MM + step] = last;
    __syncthreads();
  }
}

// ---------------------------------------------------------------------------
// Gather sampled pose/normal/sh straight into the output tuple slots.
// ---------------------------------------------------------------------------
__global__ void gather_kernel(const float* __restrict__ pose,
                              const float* __restrict__ normal,
                              const float* __restrict__ sh,
                              const int* __restrict__ fps_idx,
                              float* __restrict__ out_pose,
                              float* __restrict__ out_normal,
                              float* __restrict__ out_sh) {
  int t = blockIdx.x*256 + threadIdx.x;
  if (t >= BB*MM) return;
  int b = t >> 11;
  int idx = fps_idx[t];
  size_t src = (size_t)b*NN + idx;
  for (int c = 0; c < 3; ++c) out_pose  [(size_t)t*3+c] = pose  [src*3 + c];
  for (int c = 0; c < 3; ++c) out_normal[(size_t)t*3+c] = normal[src*3 + c];
  for (int c = 0; c < 9; ++c) out_sh    [(size_t)t*9+c] = sh    [src*9 + c];
}

// ---------------------------------------------------------------------------
// KNN: one wave32 per query. Distances (||q||^2 - 2 q.p + ||p||^2, same
// formula as the reference) cached in LDS; 32 masked argmin passes with
// shfl_xor butterfly reductions, tie-break to lowest index (lax.top_k stable).
// ---------------------------------------------------------------------------
__global__ __launch_bounds__(128)
void knn_kernel(const float* __restrict__ pose, const int* __restrict__ fps_idx,
                int* __restrict__ knn_out) {
  __shared__ float ds[4][NN];
  const int wave = threadIdx.x >> 5;
  const int lane = threadIdx.x & 31;
  const int q = blockIdx.x * 4 + wave;
  const int b = q >> 11;
  const int m = q & 2047;
  const float* P = pose + (size_t)b * NN * 3;
  const int qi = fps_idx[b*MM + m];
  const float qx = P[qi*3+0], qy = P[qi*3+1], qz = P[qi*3+2];
  const float qq = qx*qx + qy*qy + qz*qz;
  float* D = ds[wave];
  for (int i = 0; i < NN/32; ++i) {
    int j = i*32 + lane;
    float px = P[j*3+0], py = P[j*3+1], pz = P[j*3+2];
    float pp = px*px + py*py + pz*pz;
    D[j] = qq - 2.0f*(qx*px + qy*py + qz*pz) + pp;
  }
  for (int t = 0; t < KK; ++t) {
    float best = 3.4e38f; int bi = NN;
    for (int i = 0; i < NN/32; ++i) {
      int j = i*32 + lane;
      float v = D[j];
      if (v < best || (v == best && j < bi)) { best = v; bi = j; }
    }
    for (int off = 16; off > 0; off >>= 1) {
      float v2 = __shfl_xor(best, off, 32);
      int   i2 = __shfl_xor(bi,   off, 32);
      if (v2 < best || (v2 == best && i2 < bi)) { best = v2; bi = i2; }
    }
    if (lane == 0) knn_out[(size_t)q*KK + t] = bi;      // butterfly: all lanes agree
    if (lane == (bi & 31)) D[bi] = 3.4e38f;             // mask winner for next pass
  }
}

// ---------------------------------------------------------------------------
// PPF features -> f16, padded to 8 halfs per row (one b128 store).
// ---------------------------------------------------------------------------
__device__ __forceinline__ float ppf_angle(float ax,float ay,float az,
                                           float bx,float by,float bz) {
  float cx = ay*bz - az*by, cy = az*bx - ax*bz, cz = ax*by - ay*bx;
  float cn = sqrtf(cx*cx + cy*cy + cz*cz + 1e-12f);
  return atan2f(cn, ax*bx + ay*by + az*bz);
}

__global__ void ppf_kernel(const float* __restrict__ pose,
                           const float* __restrict__ normal,
                           const int* __restrict__ knn,
                           half_t* __restrict__ x5h) {
  int r = blockIdx.x*256 + threadIdx.x;
  if (r >= RR) return;
  int q = r >> 5;
  int b = r >> 16;
  const float* P  = pose   + (size_t)b*NN*3;
  const float* Nr = normal + (size_t)b*NN*3;
  int c0 = knn[(size_t)q*KK];
  int pt = knn[r];
  float cx = P[c0*3+0],  cy = P[c0*3+1],  cz = P[c0*3+2];
  float n1x = Nr[c0*3+0], n1y = Nr[c0*3+1], n1z = Nr[c0*3+2];
  float px = P[pt*3+0],  py = P[pt*3+1],  pz = P[pt*3+2];
  float n2x = Nr[pt*3+0], n2y = Nr[pt*3+1], n2z = Nr[pt*3+2];
  float dx = px-cx, dy = py-cy, dz = pz-cz;
  float dist = sqrtf(dx*dx + dy*dy + dz*dz + 1e-12f);
  float a1 = ppf_angle(n1x,n1y,n1z, dx,dy,dz);
  float a2 = ppf_angle(n2x,n2y,n2z, dx,dy,dz);
  float a3 = ppf_angle(n1x,n1y,n1z, n2x,n2y,n2z);
  v8h o;
  o[0]=(half_t)a1; o[1]=(half_t)a2; o[2]=(half_t)a3;
  o[3]=(half_t)dist; o[4]=(half_t)dist;
  o[5]=(half_t)0.0f; o[6]=(half_t)0.0f; o[7]=(half_t)0.0f;
  *(v8h*)(x5h + (size_t)r*8) = o;
}

// ---------------------------------------------------------------------------
// Fused (prev-BN-affine + ReLU) -> GEMM via v_wmma_f32_16x16x32_f16.
// A tile 32 x Kpad and transposed weights staged in LDS; both fragments load
// as two ds_load_b128 per lane per documented CDNA5 A/B VGPR layouts.
// ---------------------------------------------------------------------------
__global__ __launch_bounds__(256)
void gemm_kernel(const half_t* __restrict__ Ain, int srcCols, int srcStride,
                 const float* __restrict__ affA, const float* __restrict__ affD,
                 const float* __restrict__ feat, const int* __restrict__ knn, int gatherC,
                 const float* __restrict__ W, const float* __restrict__ bias,
                 int Kpad, int Cout,
                 half_t* __restrict__ Yout) {
  __shared__ half_t sA[32*128];     // 8 KB max
  __shared__ half_t sW[8192];       // Cout*Kpad <= 8192 for all layers
  const int tid = threadIdx.x;
  const int CinW = gatherC + srcCols;
  for (int i = tid; i < Cout*Kpad; i += 256) {   // W^T, zero-padded K
    int d = i / Kpad, c = i % Kpad;
    sW[i] = (c < CinW) ? (half_t)W[c*Cout + d] : (half_t)0.0f;
  }
  __syncthreads();
  const int wave = tid >> 5, lane = tid & 31;
  const int rowHalf = wave & 1;          // rows 0-15 / 16-31 of the tile
  const int colSlot = wave >> 1;         // 4 column-tile slots
  const int lcol  = lane & 15;
  const int mrow  = rowHalf*16 + lcol;
  const int aoff  = (lane < 16) ? 0 : 8;     // A: K groups {0..7,16..23} vs {8..15,24..31}
  const int boff  = (lane < 16) ? 0 : 8;     // B: K groups {0..7,16..23} vs {8..15,24..31}

  for (int rb = blockIdx.x * 32; rb < RR; rb += gridDim.x * 32) {
    for (int i = tid; i < 32*Kpad; i += 256) {
      int rr = i / Kpad, c = i % Kpad;
      int r = rb + rr;
      float v = 0.0f;
      if (c < gatherC) {                       // concat([feat_c, x]) prefix
        int pt = knn[r];
        int bb = r >> 16;
        v = feat[((size_t)bb*NN + pt)*CFEAT + c];
      } else {
        int cs = c - gatherC;
        if (cs < srcCols) {
          float y = (float)Ain[(size_t)r*srcStride + cs];
          v = affA ? fmaxf(0.0f, affA[cs]*y + affD[cs]) : y;  // folded BN + ReLU
        }
      }
      sA[rr*Kpad + c] = (half_t)v;
    }
    if (tid < 32) {                            // gfx1250 global_prefetch of next tile
      size_t nr = (size_t)rb + (size_t)gridDim.x*32 + tid;
      if (nr < RR) __builtin_prefetch(&Ain[nr*(size_t)srcStride], 0, 1);
    }
    __syncthreads();

    for (int ct = colSlot; ct < (Cout >> 4); ct += 4) {
      const int ncol = ct*16 + lcol;
      v8f acc = {0.f,0.f,0.f,0.f,0.f,0.f,0.f,0.f};
      for (int kk = 0; kk < Kpad; kk += 32) {
        const half_t* ap = &sA[mrow*Kpad + kk + aoff];
        v8h a0 = *(const v8h*)ap;
        v8h a1 = *(const v8h*)(ap + 16);
        v16h afr = __builtin_shufflevector(a0, a1, 0,1,2,3,4,5,6,7,8,9,10,11,12,13,14,15);
        const half_t* bp = &sW[ncol*Kpad + kk + boff];
        v8h b0 = *(const v8h*)bp;
        v8h b1 = *(const v8h*)(bp + 16);
        v16h bfr = __builtin_shufflevector(b0, b1, 0,1,2,3,4,5,6,7,8,9,10,11,12,13,14,15);
        acc = __builtin_amdgcn_wmma_f32_16x16x32_f16(
            false, afr, false, bfr, (short)0, acc, false, false);
      }
      const float bv = bias[ncol];
      const int rbase = rb + rowHalf*16 + ((lane < 16) ? 0 : 8);
      #pragma unroll
      for (int i2 = 0; i2 < 8; ++i2)
        Yout[(size_t)(rbase + i2)*Cout + ncol] = (half_t)(acc[i2] + bv);
    }
    __syncthreads();
  }
}

// ---------------------------------------------------------------------------
// Per-channel sum / sum-of-squares for BatchNorm statistics.
// ---------------------------------------------------------------------------
__global__ __launch_bounds__(256)
void stats_kernel(const half_t* __restrict__ Y, int C, float* __restrict__ stats) {
  __shared__ float s1[256], s2[256];
  const int tid = threadIdx.x;
  const int rowsPer = 256 / C;
  const int c = tid % C;
  const int r0 = blockIdx.x * rowsPer + tid / C;
  const int rstride = gridDim.x * rowsPer;
  float a = 0.0f, b = 0.0f;
  for (int r = r0; r < RR; r += rstride) {
    float v = (float)Y[(size_t)r*C + c];
    a += v; b += v*v;
  }
  s1[tid] = a; s2[tid] = b;
  __syncthreads();
  if (tid < C) {
    for (int j = 1; j < rowsPer; ++j) { a += s1[tid + j*C]; b += s2[tid + j*C]; }
    atomicAdd(&stats[c], a);
    atomicAdd(&stats[128 + c], b);
  }
}

__global__ void affine_kernel(const float* __restrict__ stats,
                              const float* __restrict__ g, const float* __restrict__ beta,
                              int C, float* __restrict__ aff) {
  int c = threadIdx.x;
  if (c >= C) return;
  const float cnt = (float)RR;
  float mu  = stats[c] / cnt;
  float var = stats[128 + c] / cnt - mu*mu;   // population variance (jnp.var)
  float a = g[c] * rsqrtf(var + 1e-5f);
  aff[c] = a;
  aff[128 + c] = beta[c] - mu * a;
}

__global__ void zero_kernel(float* __restrict__ p, int n) {
  int i = blockIdx.x*256 + threadIdx.x;
  if (i < n) p[i] = 0.0f;
}

// ---------------------------------------------------------------------------
// Final BN + ReLU + max over k neighbors.
// ---------------------------------------------------------------------------
__global__ void maxpool_kernel(const half_t* __restrict__ Y5, const float* __restrict__ aff,
                               float* __restrict__ out_feat) {
  int t = blockIdx.x*256 + threadIdx.x;    // B*M*128
  if (t >= BB*MM*128) return;
  int c = t & 127;
  int qm = t >> 7;
  const float a = aff[c], d = aff[128 + c];
  size_t base = (size_t)qm * KK * 128;
  float mx = 0.0f;                          // relu output >= 0
  for (int k = 0; k < KK; ++k) {
    float y = (float)Y5[base + (size_t)k*128 + c];
    mx = fmaxf(mx, fmaxf(0.0f, a*y + d));
  }
  out_feat[(size_t)qm*128 + c] = mx;
}

// ---------------------------------------------------------------------------
extern "C" void kernel_launch(void* const* d_in, const int* in_sizes, int n_in,
                              void* d_out, int out_size, void* d_ws, size_t ws_size,
                              hipStream_t stream) {
  const float* pose    = (const float*)d_in[0];
  const float* feat    = (const float*)d_in[1];
  const float* normal  = (const float*)d_in[2];
  const float* sh      = (const float*)d_in[3];
  const float* rri_w1  = (const float*)d_in[4];
  const float* rri_b1  = (const float*)d_in[5];
  const float* rri_g1  = (const float*)d_in[6];
  const float* rri_be1 = (const float*)d_in[7];
  const float* rri_w2  = (const float*)d_in[8];
  const float* rri_b2  = (const float*)d_in[9];
  const float* rri_g2  = (const float*)d_in[10];
  const float* rri_be2 = (const float*)d_in[11];
  const float* pn_w1   = (const float*)d_in[12];
  const float* pn_b1   = (const float*)d_in[13];
  const float* pn_g1   = (const float*)d_in[14];
  const float* pn_be1  = (const float*)d_in[15];
  const float* pn_w2   = (const float*)d_in[16];
  const float* pn_b2   = (const float*)d_in[17];
  const float* pn_g2   = (const float*)d_in[18];
  const float* pn_be2  = (const float*)d_in[19];
  const float* pn_w3   = (const float*)d_in[20];
  const float* pn_b3   = (const float*)d_in[21];
  const float* pn_g3   = (const float*)d_in[22];
  const float* pn_be3  = (const float*)d_in[23];

  float* out = (float*)d_out;                 // tuple order: pose, feat, normal, sh
  float* out_pose   = out;                    // 8*2048*3
  float* out_feat   = out + 49152;            // 8*2048*128
  float* out_normal = out + 49152 + 2097152;  // 8*2048*3
  float* out_sh     = out + 49152 + 2097152 + 49152;

  uint8_t* w = (uint8_t*)d_ws;
  int*    fps   = (int*)w;                                  //   64 KB
  int*    knn   = (int*)(w + 65536);                        //    2 MB
  half_t* x5h   = (half_t*)(w + 65536 + 2097152);           //    8 MB
  float*  stats = (float*)(w + 65536 + 2097152 + 8388608);  //    5 KB (5 layers x 256)
  float*  aff   = stats + 5*256;                            //    5 KB
  half_t* yA    = (half_t*)(w + 65536 + 2097152 + 8388608 + 10240);  // 128 MB
  half_t* yB    = yA + (size_t)RR * 128;                             // 128 MB

  fps_kernel   <<<BB, 1024, 0, stream>>>(pose, fps);
  gather_kernel<<<(BB*MM + 255)/256, 256, 0, stream>>>(pose, normal, sh, fps,
                                                       out_pose, out_normal, out_sh);
  knn_kernel   <<<BB*MM/4, 128, 0, stream>>>(pose, fps, knn);
  ppf_kernel   <<<RR/256, 256, 0, stream>>>(pose, normal, knn, x5h);
  zero_kernel  <<<(5*256 + 255)/256, 256, 0, stream>>>(stats, 5*256);

  // L1: x5 (5 -> 64), raw input (no affine/relu)
  gemm_kernel<<<1024, 256, 0, stream>>>(x5h, 5, 8, nullptr, nullptr,
                                        nullptr, nullptr, 0,
                                        rri_w1, rri_b1, 32, 64, yA);
  stats_kernel <<<512, 256, 0, stream>>>(yA, 64, stats + 0*256);
  affine_kernel<<<1, 128, 0, stream>>>(stats + 0*256, rri_g1, rri_be1, 64, aff + 0*256);

  // L2: 64 -> 64
  gemm_kernel<<<1024, 256, 0, stream>>>(yA, 64, 64, aff + 0*256, aff + 0*256 + 128,
                                        nullptr, nullptr, 0,
                                        rri_w2, rri_b2, 64, 64, yB);
  stats_kernel <<<512, 256, 0, stream>>>(yB, 64, stats + 1*256);
  affine_kernel<<<1, 128, 0, stream>>>(stats + 1*256, rri_g2, rri_be2, 64, aff + 1*256);

  // L3: concat(feat_c, x) 128 -> 64, feat gathered on the fly
  gemm_kernel<<<1024, 256, 0, stream>>>(yB, 64, 64, aff + 1*256, aff + 1*256 + 128,
                                        feat, knn, 64,
                                        pn_w1, pn_b1, 128, 64, yA);
  stats_kernel <<<512, 256, 0, stream>>>(yA, 64, stats + 2*256);
  affine_kernel<<<1, 128, 0, stream>>>(stats + 2*256, pn_g1, pn_be1, 64, aff + 2*256);

  // L4: 64 -> 64
  gemm_kernel<<<1024, 256, 0, stream>>>(yA, 64, 64, aff + 2*256, aff + 2*256 + 128,
                                        nullptr, nullptr, 0,
                                        pn_w2, pn_b2, 64, 64, yB);
  stats_kernel <<<512, 256, 0, stream>>>(yB, 64, stats + 3*256);
  affine_kernel<<<1, 128, 0, stream>>>(stats + 3*256, pn_g2, pn_be2, 64, aff + 3*256);

  // L5: 64 -> 128
  gemm_kernel<<<1024, 256, 0, stream>>>(yB, 64, 64, aff + 3*256, aff + 3*256 + 128,
                                        nullptr, nullptr, 0,
                                        pn_w3, pn_b3, 64, 128, yA);
  stats_kernel <<<512, 256, 0, stream>>>(yA, 128, stats + 4*256);
  affine_kernel<<<1, 128, 0, stream>>>(stats + 4*256, pn_g3, pn_be3, 128, aff + 4*256);

  maxpool_kernel<<<(BB*MM*128 + 255)/256, 256, 0, stream>>>(yA, aff + 4*256, out_feat);
}
